// ProtoFormer_46634754900257
// MI455X (gfx1250) — matrix operand order repaired and checked by reading, not compiled, not run on hardware
//
#include <hip/hip_runtime.h>
#include <cmath>

// ============================================================================
// ProtoFormer forward for MI455X (gfx1250, wave32, WMMA).
// All 1x1 convs -> bf16 WMMA GEMM (fp32 accumulate) with fused epilogues.
// ============================================================================

#define THREADS 256

typedef __attribute__((ext_vector_type(16))) __bf16 v16bf;
typedef __attribute__((ext_vector_type(8)))  float  v8f;

union FragU { uint4 u[2]; v16bf v; };

__device__ __forceinline__ unsigned short f2bf(float f) {
  unsigned u = __float_as_uint(f);
  unsigned r = u + 0x7fffu + ((u >> 16) & 1u);   // round-to-nearest-even
  return (unsigned short)(r >> 16);
}
__device__ __forceinline__ unsigned pack_bf2(float lo, float hi) {
  return (unsigned)f2bf(lo) | ((unsigned)f2bf(hi) << 16);
}

// ----------------------------------------------------------------------------
// GEMM: Y[b] = act( (A @ X[b] + bias) * (gamma*BNF) + beta ) * postScale + resid
//   A: [M,K] fp32 (weights, shared across batch), X: [B,K,N], Y: [B,M,N]
// Block: 256 thr = 8 waves; block tile 64x64; each wave: 16x32 C
// (one A fragment feeds two B fragments -> 2 WMMA per K step per wave).
// ----------------------------------------------------------------------------
__global__ void __launch_bounds__(256)
k_gemm(const float* __restrict__ A, const float* __restrict__ X,
       float* __restrict__ Y,
       const float* __restrict__ bias, const float* __restrict__ gam,
       const float* __restrict__ bet, const float* __restrict__ resid,
       int M, int K, int N, int act, float postScale, int inRelu)
{
  __shared__ alignas(16) unsigned short As[64 * 32];   // [m][k] bf16
  __shared__ alignas(16) unsigned short Bs[64 * 32];   // [n][k] bf16 (K-contig)

  const int tid  = threadIdx.x;
  const int z    = blockIdx.z;
  const float* Xb = X + (size_t)z * K * N;
  float*       Yb = Y + (size_t)z * M * N;
  const float* Rb = resid ? resid + (size_t)z * M * N : nullptr;

  const int mBase = blockIdx.y * 64, nBase = blockIdx.x * 64;
  const int wave = tid >> 5, lane = tid & 31;
  const int wm = wave & 3;       // 4 M-subtiles of 16 rows
  const int wn = wave >> 2;      // 2 N-subtiles of 32 cols
  const int half = lane >> 4, col = lane & 15;

  v8f acc0 = {}, acc1 = {};

  for (int k0 = 0; k0 < K; k0 += 32) {
    const bool fullK = (k0 + 32 <= K);
    // ---- stage A tile: rows [mBase,mBase+64), k [k0,k0+32) ----
    if (fullK && mBase + 64 <= M) {
      // wave covers 8 rows; 8 lanes x float4 per row; packed ds_store_b64
      const int lr = lane >> 3, lc = lane & 7;
#pragma unroll
      for (int rep = 0; rep < 2; ++rep) {
        int row = wave * 8 + rep * 4 + lr;
        const float4 f =
            *(const float4*)&A[(size_t)(mBase + row) * K + k0 + lc * 4];
        float x0 = f.x, x1 = f.y, x2 = f.z, x3 = f.w;
        uint2 p; p.x = pack_bf2(x0, x1); p.y = pack_bf2(x2, x3);
        *(uint2*)&As[row * 32 + lc * 4] = p;
      }
    } else {
      for (int i = tid; i < 64 * 32; i += 256) {
        int mm = i >> 5, kk = i & 31;
        int gm = mBase + mm, gk = k0 + kk;
        int sm = gm < M ? gm : M - 1, sk = gk < K ? gk : K - 1;
        float v = A[(size_t)sm * K + sk];            // always in-bounds
        v = (gm < M && gk < K) ? v : 0.0f;           // select, no branch
        As[i] = f2bf(v);
      }
    }
    // ---- stage B tile transposed to [n][k]: cols [nBase,nBase+64) ----
    if (fullK && nBase + 64 <= N) {
      // wave w: column group (w&1)*32 + lane, k run (w>>1)*8..+7.
      const int gn = nBase + (wave & 1) * 32 + lane;
      const int kOff = (wave >> 1) * 8;
      float t[8];
#pragma unroll
      for (int j = 0; j < 8; ++j) {
        float v = Xb[(size_t)(k0 + kOff + j) * N + gn];  // coalesced per j
        t[j] = inRelu ? fmaxf(v, 0.0f) : v;
      }
      uint4 p;
      p.x = pack_bf2(t[0], t[1]); p.y = pack_bf2(t[2], t[3]);
      p.z = pack_bf2(t[4], t[5]); p.w = pack_bf2(t[6], t[7]);
      *(uint4*)&Bs[((wave & 1) * 32 + lane) * 32 + kOff] = p;
    } else {
      for (int i = tid; i < 64 * 32; i += 256) {
        int kk = i >> 6, nn = i & 63;                // n fastest -> coalesced
        int gn = nBase + nn, gk = k0 + kk;
        int sn = gn < N ? gn : N - 1, sk = gk < K ? gk : K - 1;
        float v = Xb[(size_t)sk * N + sn];
        v = (gn < N && gk < K) ? v : 0.0f;
        if (inRelu) v = fmaxf(v, 0.0f);
        Bs[(nn << 5) + kk] = f2bf(v);
      }
    }
    if (k0 + 32 < K) {  // speculative prefetch of next K tile
      __builtin_prefetch(Xb + (size_t)(k0 + 32) * N + nBase + (tid & 31), 0, 1);
      __builtin_prefetch(A + (size_t)(mBase + (tid >> 2)) * K + k0 + 32, 0, 1);
    }
    __syncthreads();

    // A fragment (16-bit 16x32): lane half h -> K run
    // [16h,16h+15] contiguous in LDS (matches B).
    FragU fa, fb0, fb1;
    const unsigned short* arow = &As[((wm << 4) + col) << 5];
    fa.u[0] = *(const uint4*)(arow + (half << 4));
    fa.u[1] = *(const uint4*)(arow + (half << 4) + 8);
    // B fragments: lane col n, K run [16h,16h+15] contiguous (Bs is [n][k]).
    const unsigned short* brow0 = &Bs[((wn << 5) + col) << 5];
    fb0.u[0] = *(const uint4*)(brow0 + (half << 4));
    fb0.u[1] = *(const uint4*)(brow0 + (half << 4) + 8);
    const unsigned short* brow1 = &Bs[((wn << 5) + 16 + col) << 5];
    fb1.u[0] = *(const uint4*)(brow1 + (half << 4));
    fb1.u[1] = *(const uint4*)(brow1 + (half << 4) + 8);

    acc0 = __builtin_amdgcn_wmma_f32_16x16x32_bf16(
        false, fa.v, false, fb0.v, (short)0, acc0, false, false);
    acc1 = __builtin_amdgcn_wmma_f32_16x16x32_bf16(
        false, fa.v, false, fb1.v, (short)0, acc1, false, false);
    __syncthreads();
  }

  // C layout: lane l, reg r -> m = r + 8*(l/16), n = l%16
  const float BNF = 0.9999950000374997f;  // 1/sqrt(1+1e-5)
  const int gn0 = nBase + (wn << 5) + col;
  const int gn1 = gn0 + 16;
#pragma unroll
  for (int rr = 0; rr < 8; ++rr) {
    int gm = mBase + (wm << 4) + (half << 3) + rr;
    if (gm < M) {
      float b = bias ? bias[gm] : 0.0f;
      float g = gam ? gam[gm] * BNF : 1.0f;
      float be = gam ? bet[gm] : 0.0f;
      if (gn0 < N) {
        float v = acc0[rr] + b;
        v = v * g + be;
        if (act == 1)      v = fmaxf(v, 0.0f);
        else if (act == 2) v = (v >= 0.0f) ? v : 0.2f * v;
        else if (act == 3) v = tanhf(v);
        v *= postScale;
        if (Rb) v += Rb[(size_t)gm * N + gn0];
        Yb[(size_t)gm * N + gn0] = v;
      }
      if (gn1 < N) {
        float v = acc1[rr] + b;
        v = v * g + be;
        if (act == 1)      v = fmaxf(v, 0.0f);
        else if (act == 2) v = (v >= 0.0f) ? v : 0.2f * v;
        else if (act == 3) v = tanhf(v);
        v *= postScale;
        if (Rb) v += Rb[(size_t)gm * N + gn1];
        Yb[(size_t)gm * N + gn1] = v;
      }
    }
  }
}

// ---------------------------- small kernels ---------------------------------

__global__ void k_transpose(const float* __restrict__ in, float* __restrict__ out,
                            int R, int C) {
  long long i = (long long)blockIdx.x * THREADS + threadIdx.x;
  int b = blockIdx.y;
  if (i >= (long long)R * C) return;
  int r = (int)(i / C), c = (int)(i % C);
  out[((size_t)b * C + c) * R + r] = in[((size_t)b * R + r) * C + c];
}

// dst[b, cd0+c, n] = src[b, cs0+c, (srcN==1?0:n)]
__global__ void k_copy(float* __restrict__ dst, int Cd, int cd0,
                       const float* __restrict__ src, int Cs, int cs0,
                       int C, int N, int srcN) {
  long long i = (long long)blockIdx.x * THREADS + threadIdx.x;
  int b = blockIdx.y;
  if (i >= (long long)C * N) return;
  int c = (int)(i / N), n = (int)(i % N);
  dst[((size_t)b * Cd + cd0 + c) * N + n] =
      src[((size_t)b * Cs + cs0 + c) * srcN + (srcN == 1 ? 0 : n)];
}

__global__ void k_add(float* __restrict__ o, const float* __restrict__ a,
                      const float* __restrict__ b, long long n) {
  long long i = (long long)blockIdx.x * THREADS + threadIdx.x;
  if (i < n) o[i] = a[i] + b[i];
}

__global__ void k_up(const float* __restrict__ in, float* __restrict__ out,
                     int C, int n, int r) {
  long long i = (long long)blockIdx.x * THREADS + threadIdx.x;
  int b = blockIdx.y;
  long long nr = (long long)n * r;
  if (i >= (long long)C * nr) return;
  int c = (int)(i / nr); long long j = i % nr;
  out[((size_t)b * C + c) * nr + j] = in[((size_t)b * C + c) * n + (int)(j / r)];
}

__global__ void k_maxk(const float* __restrict__ in, float* __restrict__ out,
                       int C, int M, int K) {
  long long i = (long long)blockIdx.x * THREADS + threadIdx.x;
  int b = blockIdx.y;
  if (i >= (long long)C * M) return;
  int c = (int)(i / M), m = (int)(i % M);
  const float* p = in + (((size_t)b * C + c) * M + m) * K;
  float v = p[0];
  for (int k = 1; k < K; ++k) v = fmaxf(v, p[k]);
  out[((size_t)b * C + c) * M + m] = v;
}

__global__ void k_maxn(const float* __restrict__ in, float* __restrict__ out,
                       int C, int N) {
  int i = blockIdx.x * THREADS + threadIdx.x;
  int b = blockIdx.y;
  if (i >= C) return;
  const float* p = in + ((size_t)b * C + i) * N;
  float v = p[0];
  for (int n = 1; n < N; ++n) v = fmaxf(v, p[n]);
  out[(size_t)b * C + i] = v;
}

__global__ void k_softmax16(float* __restrict__ a, long long rows) {
  long long i = (long long)blockIdx.x * THREADS + threadIdx.x;
  if (i >= rows) return;
  float* p = a + i * 16;
  float m = p[0];
#pragma unroll
  for (int k = 1; k < 16; ++k) m = fmaxf(m, p[k]);
  float s = 0.f;
#pragma unroll
  for (int k = 0; k < 16; ++k) { float e = __expf(p[k] - m); p[k] = e; s += e; }
  float inv = 1.0f / s;
#pragma unroll
  for (int k = 0; k < 16; ++k) p[k] *= inv;
}

// KNN-16 smallest sqdist; both Q [B,C,M] and S [B,C,N] channel-major.
template <int C>
__global__ void k_knn(const float* __restrict__ Q, const float* __restrict__ S,
                      int* __restrict__ idx, int M, int N) {
  int i = blockIdx.x * THREADS + threadIdx.x;
  int b = blockIdx.y;
  if (i >= M) return;
  float qv[C];
#pragma unroll
  for (int c = 0; c < C; ++c) qv[c] = Q[((size_t)b * C + c) * M + i];
  float bd[16]; int bi[16];
#pragma unroll
  for (int j = 0; j < 16; ++j) { bd[j] = 3.0e38f; bi[j] = 0; }
  float worst = 3.0e38f; int wslot = 0;
  for (int n = 0; n < N; ++n) {
    float d = 0.f;
#pragma unroll
    for (int c = 0; c < C; ++c) {
      float t = qv[c] - S[((size_t)b * C + c) * N + n];
      d += t * t;
    }
    if (d < worst) {
#pragma unroll
      for (int j = 0; j < 16; ++j) if (j == wslot) { bd[j] = d; bi[j] = n; }
      float w = -1.f; int sl = 0;
#pragma unroll
      for (int j = 0; j < 16; ++j) if (bd[j] > w) { w = bd[j]; sl = j; }
      worst = w; wslot = sl;
    }
  }
#pragma unroll
  for (int j = 0; j < 16; ++j) idx[((size_t)b * M + i) * 16 + j] = bi[j];
}

// FPS: pts point-major [B,N,3] -> out point-major [B,npt,3]; one block / batch.
__global__ void k_fps(const float* __restrict__ pts, float* __restrict__ out,
                      int N, int npt) {
  int b = blockIdx.x;
  pts += (size_t)b * N * 3;
  out += (size_t)b * npt * 3;
  __shared__ float dist[4096];
  __shared__ float rv[256];
  __shared__ int   ri[256];
  __shared__ int   farS;
  int t = threadIdx.x;
  for (int i = t; i < N; i += 256) dist[i] = 1e10f;
  if (t == 0) farS = 0;
  __syncthreads();
  for (int it = 0; it < npt; ++it) {
    int far = farS;
    if (t < 3) out[it * 3 + t] = pts[far * 3 + t];
    float cx = pts[far * 3], cy = pts[far * 3 + 1], cz = pts[far * 3 + 2];
    float best = -1.f; int bi = 0;
    for (int i = t; i < N; i += 256) {
      float dx = pts[i * 3] - cx, dy = pts[i * 3 + 1] - cy, dz = pts[i * 3 + 2] - cz;
      float nd = fminf(dist[i], dx * dx + dy * dy + dz * dz);
      dist[i] = nd;
      if (nd > best) { best = nd; bi = i; }
    }
    rv[t] = best; ri[t] = bi;
    __syncthreads();
    for (int st = 128; st > 0; st >>= 1) {
      if (t < st && rv[t + st] > rv[t]) { rv[t] = rv[t + st]; ri[t] = ri[t + st]; }
      __syncthreads();
    }
    if (t == 0) farS = ri[0];
    __syncthreads();
  }
}

// gather with optional center op; writes a dense chunk [B,Cout,CH,16] at c0.
// mode 0: g ; 1: g - center ; 2: center - g
__global__ void k_gather(const float* __restrict__ src, const float* __restrict__ center,
                         const int* __restrict__ idx, float* __restrict__ out,
                         int C, int N, int Mfull, int mOff, int CH, int mode,
                         int Cout, int c0) {
  long long i = (long long)blockIdx.x * THREADS + threadIdx.x;
  int b = blockIdx.y;
  if (i >= (long long)C * CH * 16) return;
  int k = (int)(i & 15);
  long long t = i >> 4;
  int m = (int)(t % CH), c = (int)(t / CH);
  int id = idx[((size_t)b * Mfull + mOff + m) * 16 + k];
  float g = src[((size_t)b * C + c) * N + id];
  if (mode) {
    float ce = center[((size_t)b * C + c) * Mfull + mOff + m];
    g = (mode == 1) ? (g - ce) : (ce - g);
  }
  out[(((size_t)b * Cout + c0 + c) * CH + m) * 16 + k] = g;
}

// agg[b,c,mOff+m] = sum_k a[b,c,m,k]*(v_term + pe[b,c,m,k])
__global__ void k_agg(const float* __restrict__ a, const float* __restrict__ pe,
                      const float* __restrict__ v, const int* __restrict__ idx,
                      float* __restrict__ out, int C, int CH, int Mfull,
                      int mOff, int gathered) {
  long long i = (long long)blockIdx.x * THREADS + threadIdx.x;
  int b = blockIdx.y;
  if (i >= (long long)C * CH) return;
  int m = (int)(i % CH), c = (int)(i / CH);
  const float* ap = a + (((size_t)b * C + c) * CH + m) * 16;
  const float* pp = pe + (((size_t)b * C + c) * CH + m) * 16;
  const float* vr = v + ((size_t)b * C + c) * Mfull;
  const int*  ix = idx + ((size_t)b * Mfull + mOff + m) * 16;
  float s = 0.f;
#pragma unroll
  for (int k = 0; k < 16; ++k) {
    float vv = gathered ? vr[ix[k]] : vr[mOff + m];
    s += ap[k] * (vv + pp[k]);
  }
  out[((size_t)b * C + c) * Mfull + mOff + m] = s;
}

__global__ void k_onehot(float* __restrict__ o) {
  int i = blockIdx.x * THREADS + threadIdx.x;
  if (i >= 4 * 16 * 16) return;
  int n = i % 16, c = (i / 16) % 16;
  o[i] = (c == n) ? 1.0f : 0.0f;
}

// W[c=32][d=128][rr=r] -> W2[(d*r+rr)][c]
__global__ void k_repack(const float* __restrict__ W, float* __restrict__ W2, int r) {
  int i = blockIdx.x * THREADS + threadIdx.x;
  int tot = 32 * 128 * r;
  if (i >= tot) return;
  int rr = i % r, d = (i / r) % 128, c = i / (r * 128);
  W2[((size_t)(d * r + rr)) * 32 + c] = W[i];
}

// Yt [B,128*r,n] -> out [B,128,n*r]
__global__ void k_reorder(const float* __restrict__ Yt, float* __restrict__ out,
                          int n, int r) {
  long long i = (long long)blockIdx.x * THREADS + threadIdx.x;
  int b = blockIdx.y;
  long long nr = (long long)n * r;
  if (i >= 128 * nr) return;
  int d = (int)(i / nr); long long j = i % nr;
  int n0 = (int)(j / r), rr = (int)(j % r);
  out[((size_t)b * 128 + d) * nr + j] =
      Yt[((size_t)b * 128 * r + d * r + rr) * n + n0];
}

// ============================ host-side machinery ===========================

static inline int gsz(long long n) { return (int)((n + THREADS - 1) / THREADS); }

struct Arena {
  char* base; size_t off;
  float* f(size_t n) {
    float* p = (float*)(base + off);
    off += (n * sizeof(float) + 255) & ~(size_t)255;
    return p;
  }
  int* ip(size_t n) {
    int* p = (int*)(base + off);
    off += (n * sizeof(int) + 255) & ~(size_t)255;
    return p;
  }
};

struct Cv  { const float* w; const float* b; int o, i; };
struct Bnp { const float* g; const float* b; };
struct Res { Cv c1, c2, sc; };
struct Attn { Cv attn1, attn2; Bnp attn_bn; Cv pos1, pos2; Bnp pos_bn; };
struct Trf  { Attn a; Cv end, k, q, start, v; };
struct Skip { Attn a; Cv end, k, q, v; Res v_mlp; };
struct Pgt {
  Cv l1[2], l2[2], ld[2]; Res ldf; Cv lps[2]; const float* lpsw; Skip lskip;
  Cv pd[2];
  Cv x1[2], x2[2], xd[2]; Res xdf; Cv xps[2]; const float* xpsw; Skip xskip;
};

struct Reader {
  void* const* d_in; int pos;
  const float* next() { return (const float*)d_in[pos++]; }
};
static Cv rdConv(Reader& r, int o, int i) { Cv c; c.w = r.next(); c.b = r.next(); c.o = o; c.i = i; return c; }
static Bnp rdBn(Reader& r) { Bnp b; b.g = r.next(); b.b = r.next(); return b; }
static Res rdRes(Reader& r, int i, int h, int o) {  // keys: c1, c2, sc
  Res x; x.c1 = rdConv(r, h, i); x.c2 = rdConv(r, o, h); x.sc = rdConv(r, o, i); return x;
}
// trf keys sorted: attn1,attn2,attn_bn,end,k,pos1,pos2,pos_bn,q,start,v
static Trf rdTrf(Reader& r, int inC, int dim) {
  Trf t;
  t.a.attn1 = rdConv(r, dim * 4, dim);
  t.a.attn2 = rdConv(r, dim, dim * 4);
  t.a.attn_bn = rdBn(r);
  t.end = rdConv(r, inC, dim);
  t.k = rdConv(r, dim, dim);
  t.a.pos1 = rdConv(r, 64, 3);
  t.a.pos2 = rdConv(r, dim, 64);
  t.a.pos_bn = rdBn(r);
  t.q = rdConv(r, dim, dim);
  t.start = rdConv(r, dim, inC);
  t.v = rdConv(r, dim, dim);
  return t;
}
// skip keys sorted: attn1,attn2,attn_bn,end,k,pos1,pos2,pos_bn,q,v,v_mlp
static Skip rdSkip(Reader& r, int inC, int posC, int dim) {
  Skip s;
  s.a.attn1 = rdConv(r, dim * 4, dim);
  s.a.attn2 = rdConv(r, dim, dim * 4);
  s.a.attn_bn = rdBn(r);
  s.end = rdConv(r, inC, dim);
  s.k = rdConv(r, dim, inC);
  s.a.pos1 = rdConv(r, 64, posC);
  s.a.pos2 = rdConv(r, dim, 64);
  s.a.pos_bn = rdBn(r);
  s.q = rdConv(r, dim, inC);
  s.v = rdConv(r, dim, inC);
  s.v_mlp = rdRes(r, inC * 2, inC, inC);
  return s;
}
static Pgt rdPgt(Reader& r) {  // keys sorted as documented
  Pgt p;
  p.l1[0] = rdConv(r, 64, 16);   p.l1[1] = rdConv(r, 128, 64);
  p.l2[0] = rdConv(r, 256, 272); p.l2[1] = rdConv(r, 128, 256);
  p.ld[0] = rdConv(r, 64, 128);  p.ld[1] = rdConv(r, 16, 64);
  p.ldf = rdRes(r, 256, 128, 128);
  p.lps[0] = rdConv(r, 64, 128); p.lps[1] = rdConv(r, 32, 64);
  p.lpsw = r.next();
  p.lskip = rdSkip(r, 128, 16, 64);
  p.pd[0] = rdConv(r, 64, 128);  p.pd[1] = rdConv(r, 16, 64);
  p.x1[0] = rdConv(r, 64, 3);    p.x1[1] = rdConv(r, 128, 64);
  p.x2[0] = rdConv(r, 256, 1296); p.x2[1] = rdConv(r, 128, 256);
  p.xd[0] = rdConv(r, 64, 128);  p.xd[1] = rdConv(r, 3, 64);
  p.xdf = rdRes(r, 256, 128, 128);
  p.xps[0] = rdConv(r, 64, 128); p.xps[1] = rdConv(r, 32, 64);
  p.xpsw = r.next();
  p.xskip = rdSkip(r, 128, 3, 64);
  return p;
}

struct Ep {
  const float* bias = nullptr;
  const float* gam = nullptr;
  const float* bet = nullptr;
  int act = 0;            // 0 none, 1 relu, 2 leaky0.2, 3 tanh
  float ps = 1.0f;
  const float* resid = nullptr;
  int inRelu = 0;
};

static void gemm(hipStream_t s, const float* W, int M, int K, const float* X,
                 int N, float* Y, Ep e) {
  dim3 grid((N + 63) / 64, (M + 63) / 64, 4);
  k_gemm<<<grid, 256, 0, s>>>(W, X, Y, e.bias, e.gam, e.bet, e.resid,
                              M, K, N, e.act, e.ps, e.inRelu);
}
static void gemmc(hipStream_t s, const Cv& c, const float* X, int N, float* Y, Ep e) {
  e.bias = c.b;
  gemm(s, c.w, c.o, c.i, X, N, Y, e);
}

static void cpy(hipStream_t s, float* dst, int Cd, int cd0, const float* src,
                int Cs, int cs0, int C, int N, int srcN) {
  k_copy<<<dim3(gsz((long long)C * N), 4), THREADS, 0, s>>>(dst, Cd, cd0, src, Cs, cs0, C, N, srcN);
}

static void knn(hipStream_t s, const float* Q, const float* Sp, int* idx,
                int C, int M, int N) {
  dim3 g(gsz(M), 4);
  if (C == 3) k_knn<3><<<g, THREADS, 0, s>>>(Q, Sp, idx, M, N);
  else        k_knn<16><<<g, THREADS, 0, s>>>(Q, Sp, idx, M, N);
}

static float* mlp2(hipStream_t s, Arena& ws, const Cv* L, const float* x, int n,
                   int inRelu = 0) {
  float* t = ws.f((size_t)4 * L[0].o * n);
  Ep a; a.act = 1; a.inRelu = inRelu;
  gemmc(s, L[0], x, n, t, a);
  float* o = ws.f((size_t)4 * L[1].o * n);
  Ep b;
  gemmc(s, L[1], t, n, o, b);
  return o;
}

static float* mlp_res_f(hipStream_t s, Arena& ws, const Res& R, const float* x,
                        int n, float* dst) {
  Ep e1; e1.act = 1;
  float* t = ws.f((size_t)4 * R.c1.o * n);
  gemmc(s, R.c1, x, n, t, e1);
  Ep e2;
  float* sc = ws.f((size_t)4 * R.sc.o * n);
  gemmc(s, R.sc, x, n, sc, e2);
  if (!dst) dst = ws.f((size_t)4 * R.c2.o * n);
  Ep e3; e3.resid = sc;
  gemmc(s, R.c2, t, n, dst, e3);
  return dst;
}

// pos-attention core, m-chunked so everything stays L2-resident.
static void attention(hipStream_t s, Arena& ws, const Attn& A, int posC,
                      const float* pos, const int* idx, const float* q,
                      const float* kk, const float* v, int n, int gathered,
                      float* agg) {
  const int dim = 64;
  int CH = n < 256 ? n : 256;
  float* qk  = ws.f((size_t)4 * dim * CH * 16);
  float* pr  = ws.f((size_t)4 * posC * CH * 16);
  float* pe1 = ws.f((size_t)4 * 64 * CH * 16);      // pos1 out; reused for attn2 out
  float* pe  = ws.f((size_t)4 * dim * CH * 16);
  float* t4  = ws.f((size_t)4 * dim * 4 * CH * 16);
  for (int m0 = 0; m0 < n; m0 += CH) {
    int ch = (n - m0 < CH) ? (n - m0) : CH;
    int Nc = ch * 16;
    k_gather<<<dim3(gsz((long long)dim * Nc), 4), THREADS, 0, s>>>(
        kk, q, idx, qk, dim, n, n, m0, ch, 2, dim, 0);           // q - k_g
    k_gather<<<dim3(gsz((long long)posC * Nc), 4), THREADS, 0, s>>>(
        pos, pos, idx, pr, posC, n, n, m0, ch, 2, posC, 0);      // pos - pos_g
    Ep e1; e1.gam = A.pos_bn.g; e1.bet = A.pos_bn.b; e1.act = 1;
    gemmc(s, A.pos1, pr, Nc, pe1, e1);
    Ep e2;
    gemmc(s, A.pos2, pe1, Nc, pe, e2);
    k_add<<<gsz(4LL * dim * Nc), THREADS, 0, s>>>(qk, qk, pe, 4LL * dim * Nc);
    Ep e3; e3.gam = A.attn_bn.g; e3.bet = A.attn_bn.b; e3.act = 1;
    gemmc(s, A.attn1, qk, Nc, t4, e3);
    Ep e4;
    gemmc(s, A.attn2, t4, Nc, pe1, e4);
    k_softmax16<<<gsz(4LL * dim * ch), THREADS, 0, s>>>(pe1, 4LL * dim * ch);
    k_agg<<<dim3(gsz((long long)dim * ch), 4), THREADS, 0, s>>>(
        pe1, pe, v, idx, agg, dim, ch, n, m0, gathered);
  }
}

static void run_transformer(hipStream_t s, Arena& ws, const Trf& T, float* x,
                            const float* pos, int n) {
  const int dim = 64;
  Ep e0;
  float* h = ws.f((size_t)4 * dim * n);  gemmc(s, T.start, x, n, h, e0);
  int* idx = ws.ip((size_t)4 * n * 16);
  knn(s, pos, pos, idx, 3, n, n);
  float* q  = ws.f((size_t)4 * dim * n); gemmc(s, T.q, h, n, q, e0);
  float* kk = ws.f((size_t)4 * dim * n); gemmc(s, T.k, h, n, kk, e0);
  float* v  = ws.f((size_t)4 * dim * n); gemmc(s, T.v, h, n, v, e0);
  float* agg = ws.f((size_t)4 * dim * n);
  attention(s, ws, T.a, 3, pos, idx, q, kk, v, n, 0, agg);
  Ep ee; ee.resid = x;
  gemmc(s, T.end, agg, n, x, ee);      // in-place residual (1:1 read->write)
}

static float* run_skip(hipStream_t s, Arena& ws, const Skip& S, const float* pos,
                       int posC, const float* key, const float* query, int n) {
  const int dim = 64, inC = 128;
  float* cat = ws.f((size_t)4 * 256 * n);
  cpy(s, cat, 256, 0, key, 128, 0, 128, n, n);
  cpy(s, cat, 256, 128, query, 128, 0, 128, n, n);
  float* value = mlp_res_f(s, ws, S.v_mlp, cat, n, nullptr);
  Ep e0;
  float* kk = ws.f((size_t)4 * dim * n); gemmc(s, S.k, key, n, kk, e0);
  float* q  = ws.f((size_t)4 * dim * n); gemmc(s, S.q, query, n, q, e0);
  float* v  = ws.f((size_t)4 * dim * n); gemmc(s, S.v, value, n, v, e0);
  int* idx = ws.ip((size_t)4 * n * 16);
  knn(s, pos, pos, idx, posC, n, n);
  float* agg = ws.f((size_t)4 * dim * n);
  attention(s, ws, S.a, posC, pos, idx, q, kk, v, n, 1, agg);
  float* out = ws.f((size_t)4 * inC * n);
  Ep ee; ee.resid = value;
  gemmc(s, S.end, agg, n, out, ee);
  return out;
}

static void sa_module(hipStream_t s, Arena& ws, const float* xyz_cm,
                      const float* pts_cm, int N, int Cp, int Mpt,
                      const Cv& c1, const Cv& c2, float* new_xyz_cm,
                      float* out_pts, const float* xyz_pm_in) {
  const float* xyz_pm = xyz_pm_in;
  if (!xyz_pm) {
    float* t = ws.f((size_t)4 * N * 3);
    k_transpose<<<dim3(gsz((long long)3 * N), 4), THREADS, 0, s>>>(xyz_cm, t, 3, N);
    xyz_pm = t;
  }
  float* new_pm = ws.f((size_t)4 * Mpt * 3);
  k_fps<<<4, THREADS, 0, s>>>(xyz_pm, new_pm, N, Mpt);
  k_transpose<<<dim3(gsz((long long)Mpt * 3), 4), THREADS, 0, s>>>(new_pm, new_xyz_cm, Mpt, 3);
  int* idx = ws.ip((size_t)4 * Mpt * 16);
  knn(s, new_xyz_cm, xyz_cm, idx, 3, Mpt, N);
  int Cin = 3 + Cp;
  float* gb = ws.f((size_t)4 * Cin * Mpt * 16);
  k_gather<<<dim3(gsz((long long)3 * Mpt * 16), 4), THREADS, 0, s>>>(
      xyz_cm, new_xyz_cm, idx, gb, 3, N, Mpt, 0, Mpt, 1, Cin, 0);     // g - center
  k_gather<<<dim3(gsz((long long)Cp * Mpt * 16), 4), THREADS, 0, s>>>(
      pts_cm, nullptr, idx, gb, Cp, N, Mpt, 0, Mpt, 0, Cin, 3);
  float* h1 = ws.f((size_t)4 * c1.o * Mpt * 16);
  Ep e1; e1.act = 1; gemmc(s, c1, gb, Mpt * 16, h1, e1);
  float* h2 = ws.f((size_t)4 * c2.o * Mpt * 16);
  Ep e2; gemmc(s, c2, h1, Mpt * 16, h2, e2);
  k_maxk<<<dim3(gsz((long long)c2.o * Mpt), 4), THREADS, 0, s>>>(h2, out_pts, c2.o, Mpt, 16);
}

static void pgt_stage(hipStream_t s, Arena& ws, const Pgt& P, int r, int i, int n,
                      const float* feat_g, const float* pl_in, const float* proto_in,
                      const float* k_prev, float* pl_out, float* proto_out,
                      float* k_out) {
  int nr = n * r;
  float* xyz = ws.f((size_t)4 * 3 * n);
  cpy(s, xyz, 3, 0, pl_in, 19, 0, 3, n, n);
  float* lab = ws.f((size_t)4 * 16 * n);
  cpy(s, lab, 16, 0, pl_in, 19, 3, 16, n, n);
  // --- xyz branch ---
  float* fx = mlp2(s, ws, P.x1, xyz, n);
  float* gmax = ws.f(4 * 128);
  k_maxn<<<dim3(gsz(128), 4), THREADS, 0, s>>>(fx, gmax, 128, n);
  float* cat = ws.f((size_t)4 * 1296 * n);
  cpy(s, cat, 1296, 0, fx, 128, 0, 128, n, n);
  cpy(s, cat, 1296, 128, gmax, 128, 0, 128, n, 1);
  cpy(s, cat, 1296, 256, proto_in, 16, 0, 16, n, n);
  cpy(s, cat, 1296, 272, feat_g, 1024, 0, 1024, n, 1);
  float* qx = mlp2(s, ws, P.x2, cat, n);
  const float* key0 = k_prev ? k_prev : qx;
  float* vxyz = run_skip(s, ws, P.xskip, xyz, 3, key0, qx, n);
  float* ps = mlp2(s, ws, P.xps, vxyz, n);
  float* W2 = ws.f((size_t)128 * r * 32);
  k_repack<<<gsz(32 * 128 * r), THREADS, 0, s>>>(P.xpsw, W2, r);
  float* yt = ws.f((size_t)4 * 128 * r * n);
  Ep ept; gemm(s, W2, 128 * r, 32, ps, n, yt, ept);
  float* fv = ws.f((size_t)4 * 128 * nr);
  k_reorder<<<dim3(gsz(128LL * nr), 4), THREADS, 0, s>>>(yt, fv, n, r);
  float* upv = ws.f((size_t)4 * 128 * nr);
  k_up<<<dim3(gsz(128LL * nr), 4), THREADS, 0, s>>>(vxyz, upv, 128, n, r);
  float* cat2 = ws.f((size_t)4 * 256 * nr);
  cpy(s, cat2, 256, 0, fv, 128, 0, 128, nr, nr);
  cpy(s, cat2, 256, 128, upv, 128, 0, 128, nr, nr);
  float* kc = mlp_res_f(s, ws, P.xdf, cat2, nr, nullptr);
  float* d1 = ws.f((size_t)4 * 64 * nr);
  Ep ed; ed.act = 1; ed.inRelu = 1; gemmc(s, P.xd[0], kc, nr, d1, ed);
  float* dx = ws.f((size_t)4 * 3 * nr);
  Ep et; et.act = 3; et.ps = 1.0f / powf(1.2f, (float)i);
  gemmc(s, P.xd[1], d1, nr, dx, et);
  float* upx = ws.f((size_t)4 * 3 * nr);
  k_up<<<dim3(gsz(3LL * nr), 4), THREADS, 0, s>>>(xyz, upx, 3, n, r);
  float* xyzf = ws.f((size_t)4 * 3 * nr);
  k_add<<<gsz(4LL * 3 * nr), THREADS, 0, s>>>(xyzf, upx, dx, 4LL * 3 * nr);
  // --- label branch ---
  float* fl = mlp2(s, ws, P.l1, lab, n);
  float* lmax = ws.f(4 * 128);
  k_maxn<<<dim3(gsz(128), 4), THREADS, 0, s>>>(fl, lmax, 128, n);
  float* catl = ws.f((size_t)4 * 272 * n);
  cpy(s, catl, 272, 0, fl, 128, 0, 128, n, n);
  cpy(s, catl, 272, 128, lmax, 128, 0, 128, n, 1);
  cpy(s, catl, 272, 256, proto_in, 16, 0, 16, n, n);
  float* vl = mlp2(s, ws, P.l2, catl, n);
  float* psl = mlp2(s, ws, P.lps, vl, n);
  float* W2l = ws.f((size_t)128 * r * 32);
  k_repack<<<gsz(32 * 128 * r), THREADS, 0, s>>>(P.lpsw, W2l, r);
  float* ytl = ws.f((size_t)4 * 128 * r * n);
  Ep e5; gemm(s, W2l, 128 * r, 32, psl, n, ytl, e5);
  float* fvl0 = ws.f((size_t)4 * 128 * nr);
  k_reorder<<<dim3(gsz(128LL * nr), 4), THREADS, 0, s>>>(ytl, fvl0, n, r);
  float* labup = ws.f((size_t)4 * 16 * nr);
  k_up<<<dim3(gsz(16LL * nr), 4), THREADS, 0, s>>>(lab, labup, 16, n, r);
  float* fvl = run_skip(s, ws, P.lskip, labup, 16, fvl0, kc, nr);
  float* cat3 = ws.f((size_t)4 * 256 * nr);
  cpy(s, cat3, 256, 0, fvl, 128, 0, 128, nr, nr);
  cpy(s, cat3, 256, 128, fvl, 128, 0, 128, nr, nr);
  mlp_res_f(s, ws, P.ldf, cat3, nr, k_out);     // k_curr_
  float* dl1 = ws.f((size_t)4 * 64 * nr);
  Ep e6; e6.act = 1; e6.inRelu = 1; gemmc(s, P.ld[0], k_out, nr, dl1, e6);
  float* dl = ws.f((size_t)4 * 16 * nr);
  Ep e7; e7.act = 3; e7.ps = 1.0f / powf(0.9f, (float)i);
  gemmc(s, P.ld[1], dl1, nr, dl, e7);
  float* labf = ws.f((size_t)4 * 16 * nr);
  k_add<<<gsz(4LL * 16 * nr), THREADS, 0, s>>>(labf, labup, dl, 4LL * 16 * nr);
  // --- prototype update ---
  float* dp1 = ws.f((size_t)4 * 64 * nr);
  Ep e8; e8.act = 1; e8.inRelu = 1; gemmc(s, P.pd[0], k_out, nr, dp1, e8);
  float* dp = ws.f((size_t)4 * 16 * nr);
  Ep e9; gemmc(s, P.pd[1], dp1, nr, dp, e9);
  float* upp = ws.f((size_t)4 * 16 * nr);
  k_up<<<dim3(gsz(16LL * nr), 4), THREADS, 0, s>>>(proto_in, upp, 16, n, r);
  k_add<<<gsz(4LL * 16 * nr), THREADS, 0, s>>>(proto_out, upp, dp, 4LL * 16 * nr);
  // --- assemble pcd_label ---
  cpy(s, pl_out, 19, 0, xyzf, 3, 0, 3, nr, nr);
  cpy(s, pl_out, 19, 3, labf, 16, 0, 16, nr, nr);
}

// ================================ entry =====================================

extern "C" void kernel_launch(void* const* d_in, const int* in_sizes, int n_in,
                              void* d_out, int out_size, void* d_ws, size_t ws_size,
                              hipStream_t stream) {
  (void)in_sizes; (void)n_in; (void)out_size; (void)ws_size;
  hipStream_t s = stream;
  const float* pcd = (const float*)d_in[0];     // [4,4096,3]

  // ---- parameters (jax pytree order: sorted dict keys) ----
  Reader r{d_in, 1};
  Cv lm1 = rdConv(r, 256, 1027); Bnp lmb1 = rdBn(r);
  Cv lm2 = rdConv(r, 64, 256);   Bnp lmb2 = rdBn(r);
  Cv lm3 = rdConv(r, 16, 64);
  Pgt pg0 = rdPgt(r), pg1 = rdPgt(r);
  Cv pm1 = rdConv(r, 128, 16);  Bnp pmb1 = rdBn(r);
  Cv pm2 = rdConv(r, 256, 128); Bnp pmb2 = rdBn(r);
  Cv pm3 = rdConv(r, 2048, 256);
  Cv sa1a = rdConv(r, 64, 6),    sa1b = rdConv(r, 128, 64);
  Cv sa2a = rdConv(r, 128, 131), sa2b = rdConv(r, 256, 128);
  Cv sa3a = rdConv(r, 512, 259), sa3b = rdConv(r, 1024, 512);
  Trf tr1 = rdTrf(r, 128, 64), tr2 = rdTrf(r, 256, 64);

  Arena ws{(char*)d_ws, 0};
  // persistent buffers
  float* pc      = ws.f((size_t)4 * 3 * 4096);
  float* l1_xyz  = ws.f((size_t)4 * 3 * 512);
  float* l1_pts  = ws.f((size_t)4 * 128 * 512);
  float* l2_xyz  = ws.f((size_t)4 * 3 * 128);
  float* l2_pts  = ws.f((size_t)4 * 256 * 128);
  float* feat_g  = ws.f((size_t)4 * 1024);
  float* coarse_t = ws.f((size_t)4 * 3 * 2048);
  float* label   = ws.f((size_t)4 * 16 * 2048);
  float* proto0  = ws.f((size_t)4 * 16 * 2048);
  float* pl0     = ws.f((size_t)4 * 19 * 2048);
  float* pl1     = ws.f((size_t)4 * 19 * 2048);
  float* proto1  = ws.f((size_t)4 * 16 * 2048);
  float* kcur0   = ws.f((size_t)4 * 128 * 2048);
  float* pl2     = ws.f((size_t)4 * 19 * 4096);
  float* proto2  = ws.f((size_t)4 * 16 * 4096);
  float* kcur1   = ws.f((size_t)4 * 128 * 4096);
  size_t mark = ws.off;

  // pc = transpose(pcd)
  k_transpose<<<dim3(gsz(4096LL * 3), 4), THREADS, 0, s>>>(pcd, pc, 4096, 3);

  // SA1 + transformer1
  sa_module(s, ws, pc, pc, 4096, 3, 512, sa1a, sa1b, l1_xyz, l1_pts, pcd);
  ws.off = mark;
  run_transformer(s, ws, tr1, l1_pts, l1_xyz, 512);
  ws.off = mark;
  // SA2 + transformer2
  sa_module(s, ws, l1_xyz, l1_pts, 512, 128, 128, sa2a, sa2b, l2_xyz, l2_pts, nullptr);
  ws.off = mark;
  run_transformer(s, ws, tr2, l2_pts, l2_xyz, 128);
  ws.off = mark;
  // SA3 (global feature)
  {
    float* cat3 = ws.f((size_t)4 * 259 * 128);
    cpy(s, cat3, 259, 0, l2_xyz, 3, 0, 3, 128, 128);
    cpy(s, cat3, 259, 3, l2_pts, 256, 0, 256, 128, 128);
    float* h = ws.f((size_t)4 * 512 * 128);
    Ep e1; e1.act = 1; gemmc(s, sa3a, cat3, 128, h, e1);
    float* h2 = ws.f((size_t)4 * 1024 * 128);
    Ep e2; gemmc(s, sa3b, h, 128, h2, e2);
    k_maxn<<<dim3(gsz(1024), 4), THREADS, 0, s>>>(h2, feat_g, 1024, 128);
  }
  ws.off = mark;
  // coarse seed + label + prototype
  {
    float* coarse_pm = ws.f((size_t)4 * 2048 * 3);
    k_fps<<<4, THREADS, 0, s>>>(pcd, coarse_pm, 4096, 2048);
    k_transpose<<<dim3(gsz(2048LL * 3), 4), THREADS, 0, s>>>(coarse_pm, coarse_t, 2048, 3);
    float* lc = ws.f((size_t)4 * 1027 * 2048);
    cpy(s, lc, 1027, 0, coarse_t, 3, 0, 3, 2048, 2048);
    cpy(s, lc, 1027, 3, feat_g, 1024, 0, 1024, 2048, 1);
    float* s1 = ws.f((size_t)4 * 256 * 2048);
    Ep e1; e1.gam = lmb1.g; e1.bet = lmb1.b; e1.act = 2;
    gemmc(s, lm1, lc, 2048, s1, e1);
    float* s2 = ws.f((size_t)4 * 64 * 2048);
    Ep e2; e2.gam = lmb2.g; e2.bet = lmb2.b; e2.act = 2;
    gemmc(s, lm2, s1, 2048, s2, e2);
    Ep e3; gemmc(s, lm3, s2, 2048, label, e3);
    // prototype
    float* oh = ws.f((size_t)4 * 16 * 16);
    k_onehot<<<gsz(4 * 16 * 16), THREADS, 0, s>>>(oh);
    float* p1 = ws.f((size_t)4 * 128 * 16);
    Ep p1e; p1e.gam = pmb1.g; p1e.bet = pmb1.b; p1e.act = 2;
    gemmc(s, pm1, oh, 16, p1, p1e);
    float* p2 = ws.f((size_t)4 * 256 * 16);
    Ep p2e; p2e.gam = pmb2.g; p2e.bet = pmb2.b; p2e.act = 2;
    gemmc(s, pm2, p1, 16, p2, p2e);
    float* p3 = ws.f((size_t)4 * 2048 * 16);
    Ep p3e; gemmc(s, pm3, p2, 16, p3, p3e);
    k_transpose<<<dim3(gsz(2048LL * 16), 4), THREADS, 0, s>>>(p3, proto0, 2048, 16);
    cpy(s, pl0, 19, 0, coarse_t, 3, 0, 3, 2048, 2048);
    cpy(s, pl0, 19, 3, label, 16, 0, 16, 2048, 2048);
  }
  ws.off = mark;
  // PGT stage 0 (r=1) and stage 1 (r=2)
  pgt_stage(s, ws, pg0, 1, 0, 2048, feat_g, pl0, proto0, nullptr, pl1, proto1, kcur0);
  float* out0 = (float*)d_out;
  k_transpose<<<dim3(gsz(19LL * 2048), 4), THREADS, 0, s>>>(pl1, out0, 19, 2048);
  ws.off = mark;
  pgt_stage(s, ws, pg1, 2, 1, 2048, feat_g, pl1, proto1, kcur0, pl2, proto2, kcur1);
  float* out1 = out0 + (size_t)4 * 2048 * 19;
  k_transpose<<<dim3(gsz(19LL * 4096), 4), THREADS, 0, s>>>(pl2, out1, 19, 4096);
  (void)kcur1; (void)proto2;
}
